// MedicalAttentionLayer_64330020159738
// MI455X (gfx1250) — compile-verified, matrix operand order checked by reading, not run on hardware
//
#include <hip/hip_runtime.h>
#include <hip/hip_bf16.h>

typedef __bf16 bf16;
typedef bf16  v16bf __attribute__((ext_vector_type(16)));
typedef bf16  v8bf  __attribute__((ext_vector_type(8)));
typedef float v8f   __attribute__((ext_vector_type(8)));

#define SEQ    2048
#define DMODEL 1024
#define NHEAD  16
#define DHEAD  64
#define MTOT   4096   // B * SEQ

static __device__ __forceinline__ v16bf cat8(v8bf lo, v8bf hi) {
  return __builtin_shufflevector(lo, hi, 0,1,2,3,4,5,6,7,8,9,10,11,12,13,14,15);
}

static __device__ __forceinline__ v8f wmma_bf16(v16bf a, v16bf b, v8f c) {
  // D = A x B + C, f32 accumulate, bf16 inputs (16x16x32)
  return __builtin_amdgcn_wmma_f32_16x16x32_bf16(false, a, false, b, (short)0, c,
                                                 false, false);
}

// Fragment load per ISA 7.12.2 (16-bit A/B 16x32):
//   lanes 0-15 : K = kb+0..7  and kb+16..23
//   lanes 16-31: K = kb+8..15 and kb+24..31
static __device__ __forceinline__ v16bf ldfrag(const bf16* rowptr, int kb, int hf) {
  v8bf lo = *(const v8bf*)(rowptr + kb + hf * 8);
  v8bf hi = *(const v8bf*)(rowptr + kb + 16 + hf * 8);
  return cat8(lo, hi);
}

// ---------------------------------------------------------------- cvt fp32->bf16
__global__ __launch_bounds__(256) void cvt_bf16(const float* __restrict__ src,
                                                bf16* __restrict__ dst, int n) {
  for (int i = blockIdx.x * blockDim.x + threadIdx.x; i < n;
       i += gridDim.x * blockDim.x)
    dst[i] = (bf16)src[i];
}

// ---------------------------------------------------------------- dense GEMM
// out(m,n) = sum_k A[m][k] * W[n][k]; 128x128 tile, K staged 64 at a time into
// double-buffered LDS via CDNA5 async global->LDS DMA (ASYNCcnt).
// EPI=0: fused QKV epilogue (+bias, scatter to head layouts, V transposed)
// EPI=1: output projection epilogue (+bo, +residual, f32 out)
template <int EPI>
__global__ __launch_bounds__(256) void gemm128(
    const bf16* __restrict__ A, const bf16* __restrict__ W,
    const float* __restrict__ bia0, const float* __restrict__ bia1,
    const float* __restrict__ bia2, const float* __restrict__ hs,
    bf16* __restrict__ Qh, bf16* __restrict__ Kh, bf16* __restrict__ Vt,
    float* __restrict__ Of) {
  __shared__ bf16 As[2][128][72];
  __shared__ bf16 Bs[2][128][72];

  const int bm = blockIdx.x * 128;
  const int bn = blockIdx.y * 128;
  const int tid = threadIdx.x;
  const int wid = tid >> 5, lane = tid & 31;
  const int wm = (wid >> 2) * 64, wn = (wid & 3) * 32;
  const int hf = lane >> 4, r = lane & 15;
  const int lrow = tid & 127, lcs = (tid >> 7) * 32;

  // LDS byte address for async DMA dest: generic ptr low 32 bits == LDS offset
  const unsigned ldsA = (unsigned)(size_t)&As[0][lrow][lcs];
  const unsigned ldsB = (unsigned)(size_t)&Bs[0][lrow][lcs];
  const unsigned bufStride = (unsigned)sizeof(bf16) * 128u * 72u;  // 18432
  const unsigned aOff0 = (unsigned)(((bm + lrow) * DMODEL + lcs) * 2);
  const unsigned wOff0 = (unsigned)(((bn + lrow) * DMODEL + lcs) * 2);
  const unsigned long long aBase = (unsigned long long)(size_t)A;
  const unsigned long long wBase = (unsigned long long)(size_t)W;

  auto stage = [&](int buf, int k0) {
    const unsigned la = ldsA + (unsigned)buf * bufStride;
    const unsigned lb = ldsB + (unsigned)buf * bufStride;
    const unsigned ao = aOff0 + (unsigned)(k0 * 2);
    const unsigned wo = wOff0 + (unsigned)(k0 * 2);
#pragma unroll
    for (int c = 0; c < 4; ++c) {
      asm volatile("global_load_async_to_lds_b128 %0, %1, %2 offset:0" ::
                       "v"(la + (unsigned)c * 16u),
                   "v"(ao + (unsigned)c * 16u), "s"(aBase)
                   : "memory");
      asm volatile("global_load_async_to_lds_b128 %0, %1, %2 offset:0" ::
                       "v"(lb + (unsigned)c * 16u),
                   "v"(wo + (unsigned)c * 16u), "s"(wBase)
                   : "memory");
    }
  };

  v8f acc[4][2] = {};

  stage(0, 0);
  asm volatile("s_wait_asynccnt 0x0" ::: "memory");
  __syncthreads();

  int p = 0;
  for (int k0 = 0; k0 < DMODEL; k0 += 64) {
    if (k0 + 64 < DMODEL) stage(p ^ 1, k0 + 64);  // overlap DMA with WMMA
#pragma unroll
    for (int ks = 0; ks < 2; ++ks) {
      const int kb = ks * 32;
      v16bf af[4], bfg[2];
#pragma unroll
      for (int mi = 0; mi < 4; ++mi)
        af[mi] = ldfrag(&As[p][wm + mi * 16 + r][0], kb, hf);
#pragma unroll
      for (int ni = 0; ni < 2; ++ni)
        bfg[ni] = ldfrag(&Bs[p][wn + ni * 16 + r][0], kb, hf);
#pragma unroll
      for (int mi = 0; mi < 4; ++mi)
#pragma unroll
        for (int ni = 0; ni < 2; ++ni)
          acc[mi][ni] = wmma_bf16(af[mi], bfg[ni], acc[mi][ni]);
    }
    asm volatile("s_wait_asynccnt 0x0" ::: "memory");
    __syncthreads();
    p ^= 1;
  }

#pragma unroll
  for (int mi = 0; mi < 4; ++mi)
#pragma unroll
    for (int ni = 0; ni < 2; ++ni) {
      const int ng = bn + wn + ni * 16 + r;
      if (EPI == 0) {
        const int which = ng >> 10;  // 0=Q 1=K 2=V
        const int nl = ng & 1023;
        const int h = nl >> 6, d = nl & 63;
        const float bias = (which == 0 ? bia0 : which == 1 ? bia1 : bia2)[nl];
#pragma unroll
        for (int i = 0; i < 8; ++i) {
          const int m = bm + wm + mi * 16 + hf * 8 + i;
          const int b = m >> 11, s = m & 2047;
          const float val = acc[mi][ni][i] + bias;
          const int bh = b * NHEAD + h;
          if (which == 0)      Qh[(bh * SEQ + s) * DHEAD + d] = (bf16)val;
          else if (which == 1) Kh[(bh * SEQ + s) * DHEAD + d] = (bf16)val;
          else                 Vt[(bh * DHEAD + d) * SEQ + s] = (bf16)val;
        }
      } else {
        const float bias = bia0[ng];
#pragma unroll
        for (int i = 0; i < 8; ++i) {
          const int m = bm + wm + mi * 16 + hf * 8 + i;
          Of[m * DMODEL + ng] = acc[mi][ni][i] + bias + hs[m * DMODEL + ng];
        }
      }
    }
}

// ---------------------------------------------------------------- attention core
// One block = one (b,h) x 16-query tile. Scores strip kept f32 in big LDS
// (~132 KB -- only possible with CDNA5's 320 KB LDS/WGP).
__global__ __launch_bounds__(256) void attn(
    const bf16* __restrict__ Qh, const bf16* __restrict__ Kh,
    const bf16* __restrict__ Vt, const float* __restrict__ mask,
    const float* __restrict__ med_bias, bf16* __restrict__ Ctx) {
  __shared__ float sc[16][SEQ + 4];
  __shared__ float red[16][17];
  __shared__ float rowsum[16];
  __shared__ float ctx_part[4][256];

  const int bh = blockIdx.x;           // 0..31
  const int qt = blockIdx.y;           // 0..127
  const int b = bh >> 4, h = bh & 15;
  const int tid = threadIdx.x;
  const int wid = tid >> 5, lane = tid & 31;
  const int hf = lane >> 4, r = lane & 15;
  const float scale = 0.125f;          // 1/sqrt(64)
  const float mb = med_bias[h];

  // Q fragments for this 16-row tile (DH=64 -> 2 k-steps), loaded once.
  const bf16* Qrow = Qh + (bh * SEQ + qt * 16) * DHEAD;
  v16bf qa[2];
#pragma unroll
  for (int ks = 0; ks < 2; ++ks) qa[ks] = ldfrag(Qrow + r * DHEAD, ks * 32, hf);

  // scores = Q K^T * scale + med_bias + mask  -> LDS strip
  for (int nt = wid; nt < SEQ / 16; nt += 8) {
    const int kbase = nt * 16;
    const bf16* Krow = Kh + (bh * SEQ + kbase) * DHEAD + r * DHEAD;
    if (nt + 8 < SEQ / 16) __builtin_prefetch(Krow + 8 * 16 * DHEAD, 0, 1);
    v8f s = {};
#pragma unroll
    for (int ks = 0; ks < 2; ++ks)
      s = wmma_bf16(qa[ks], ldfrag(Krow, ks * 32, hf), s);
    const float mk = mask[b * SEQ + kbase + r];
#pragma unroll
    for (int i = 0; i < 8; ++i)
      sc[hf * 8 + i][kbase + r] = s[i] * scale + mb + mk;
  }
  __syncthreads();

  // block softmax over 2048 (16 threads per row, 128 elems each);
  // keep un-normalized exp in LDS, fold 1/sum into the context epilogue.
  {
    const int row = tid >> 4, seg = tid & 15;
    float mx = -3.0e38f;
    const int base = seg * 128;
#pragma unroll 4
    for (int j = 0; j < 128; ++j) mx = fmaxf(mx, sc[row][base + j]);
    red[row][seg] = mx;
    __syncthreads();
    if (seg == 0) {
      float m = red[row][0];
      for (int j = 1; j < 16; ++j) m = fmaxf(m, red[row][j]);
      red[row][16] = m;
    }
    __syncthreads();
    const float m = red[row][16];
    float psum = 0.f;
#pragma unroll 4
    for (int j = 0; j < 128; ++j) {
      const float e = __expf(sc[row][base + j] - m);
      sc[row][base + j] = e;
      psum += e;
    }
    red[row][seg] = psum;
    __syncthreads();
    if (seg == 0) {
      float t = 0.f;
      for (int j = 0; j < 16; ++j) t += red[row][j];
      rowsum[row] = t;
    }
    __syncthreads();
  }

  // ctx = P V : 4 d-tiles of 16, split-K across wave pairs
  const int dt = wid & 3, kh2 = wid >> 2;
  v8f cacc = {};
  const bf16* Vrow = Vt + (bh * DHEAD + dt * 16 + r) * SEQ;
  for (int kt = kh2 * 32; kt < kh2 * 32 + 32; ++kt) {
    const int kb = kt * 32;
    float tmp[16];
    const float* srow = &sc[r][0];
    *(float4*)&tmp[0]  = *(const float4*)&srow[kb + hf * 8];
    *(float4*)&tmp[4]  = *(const float4*)&srow[kb + hf * 8 + 4];
    *(float4*)&tmp[8]  = *(const float4*)&srow[kb + 16 + hf * 8];
    *(float4*)&tmp[12] = *(const float4*)&srow[kb + 16 + hf * 8 + 4];
    v16bf a;
#pragma unroll
    for (int j = 0; j < 16; ++j) a[j] = (bf16)tmp[j];
    cacc = wmma_bf16(a, ldfrag(Vrow, kb, hf), cacc);
  }
  if (kh2 == 1) {
#pragma unroll
    for (int i = 0; i < 8; ++i) ctx_part[dt][i * 32 + lane] = cacc[i];
  }
  __syncthreads();
  if (kh2 == 0) {
#pragma unroll
    for (int i = 0; i < 8; ++i) {
      const int m = hf * 8 + i;
      const float val = (cacc[i] + ctx_part[dt][i * 32 + lane]) / rowsum[m];
      Ctx[(b * SEQ + qt * 16 + m) * DMODEL + h * DHEAD + dt * 16 + r] = (bf16)val;
    }
  }
}

// ---------------------------------------------------------------- LayerNorm
__global__ __launch_bounds__(256) void lnorm(const float* __restrict__ x,
                                             const float* __restrict__ gamma,
                                             const float* __restrict__ beta,
                                             float* __restrict__ out) {
  __shared__ float rbuf[256];
  const int tid = threadIdx.x;
  const float* xr = x + (size_t)blockIdx.x * DMODEL;
  float4 v = ((const float4*)xr)[tid];
  rbuf[tid] = v.x + v.y + v.z + v.w;
  __syncthreads();
  for (int off = 128; off > 0; off >>= 1) {
    if (tid < off) rbuf[tid] += rbuf[tid + off];
    __syncthreads();
  }
  const float mu = rbuf[0] * (1.0f / DMODEL);
  __syncthreads();
  float4 d;
  d.x = v.x - mu; d.y = v.y - mu; d.z = v.z - mu; d.w = v.w - mu;
  rbuf[tid] = d.x * d.x + d.y * d.y + d.z * d.z + d.w * d.w;
  __syncthreads();
  for (int off = 128; off > 0; off >>= 1) {
    if (tid < off) rbuf[tid] += rbuf[tid + off];
    __syncthreads();
  }
  const float inv = rsqrtf(rbuf[0] * (1.0f / DMODEL) + 1e-5f);
  const float4 g = ((const float4*)gamma)[tid];
  const float4 be = ((const float4*)beta)[tid];
  float4 o;
  o.x = d.x * inv * g.x + be.x;
  o.y = d.y * inv * g.y + be.y;
  o.z = d.z * inv * g.z + be.z;
  o.w = d.w * inv * g.w + be.w;
  ((float4*)(out + (size_t)blockIdx.x * DMODEL))[tid] = o;
}

// ---------------------------------------------------------------- launcher
extern "C" void kernel_launch(void* const* d_in, const int* in_sizes, int n_in,
                              void* d_out, int out_size, void* d_ws, size_t ws_size,
                              hipStream_t stream) {
  const float* hs   = (const float*)d_in[0];
  const float* mask = (const float*)d_in[1];
  const float* Wq   = (const float*)d_in[2];
  const float* bq   = (const float*)d_in[3];
  const float* Wk   = (const float*)d_in[4];
  const float* bk   = (const float*)d_in[5];
  const float* Wv   = (const float*)d_in[6];
  const float* bv   = (const float*)d_in[7];
  const float* med  = (const float*)d_in[8];
  const float* Wo   = (const float*)d_in[9];
  const float* bo   = (const float*)d_in[10];
  const float* gam  = (const float*)d_in[11];
  const float* bet  = (const float*)d_in[12];

  char* ws = (char*)d_ws;
  size_t off = 0;
  bf16* Xbf  = (bf16*)(ws + off); off += (size_t)MTOT * DMODEL * 2;       // 8 MiB
  bf16* Wqkv = (bf16*)(ws + off); off += (size_t)3 * DMODEL * DMODEL * 2; // 6 MiB
  bf16* Wob  = (bf16*)(ws + off); off += (size_t)DMODEL * DMODEL * 2;     // 2 MiB
  bf16* Qh   = (bf16*)(ws + off); off += (size_t)MTOT * DMODEL * 2;       // 8 MiB
  bf16* Kh   = (bf16*)(ws + off); off += (size_t)MTOT * DMODEL * 2;       // 8 MiB
  bf16* Vt   = (bf16*)(ws + off); off += (size_t)MTOT * DMODEL * 2;       // 8 MiB
  bf16* Ctx  = (bf16*)(ws + off); off += (size_t)MTOT * DMODEL * 2;       // 8 MiB
  float* Xres = (float*)(ws + off);                                       // 16 MiB

  const int NW = DMODEL * DMODEL;
  cvt_bf16<<<2048, 256, 0, stream>>>(hs, Xbf, MTOT * DMODEL);
  cvt_bf16<<<1024, 256, 0, stream>>>(Wq, Wqkv, NW);
  cvt_bf16<<<1024, 256, 0, stream>>>(Wk, Wqkv + NW, NW);
  cvt_bf16<<<1024, 256, 0, stream>>>(Wv, Wqkv + 2 * NW, NW);
  cvt_bf16<<<1024, 256, 0, stream>>>(Wo, Wob, NW);

  dim3 gqkv(MTOT / 128, 3 * DMODEL / 128);    // 32 x 24
  gemm128<0><<<gqkv, 256, 0, stream>>>(Xbf, Wqkv, bq, bk, bv, nullptr,
                                       Qh, Kh, Vt, nullptr);

  dim3 gat(2 * NHEAD, SEQ / 16);              // 32 x 128
  attn<<<gat, 256, 0, stream>>>(Qh, Kh, Vt, mask, med, Ctx);

  dim3 go(MTOT / 128, DMODEL / 128);          // 32 x 8
  gemm128<1><<<go, 256, 0, stream>>>(Ctx, Wob, bo, nullptr, nullptr, hs,
                                     nullptr, nullptr, nullptr, Xres);

  lnorm<<<MTOT, 256, 0, stream>>>(Xres, gam, bet, (float*)d_out);
}
